// STNILM_13211319403309
// MI455X (gfx1250) — compile-verified, matrix operand order checked by reading, not run on hardware
//
#include <hip/hip_runtime.h>
#include <hip/hip_bf16.h>

typedef __bf16 bf16_t;
typedef __attribute__((ext_vector_type(16))) __bf16 v16bf;
typedef __attribute__((ext_vector_type(8)))  __bf16 v8bf;
typedef __attribute__((ext_vector_type(4)))  __bf16 v4bf;
typedef __attribute__((ext_vector_type(8)))  float  v8f;

#define TM 64
#define TN 64
#define TK 32
#define LDK 40            // padded LDS K-stride (80B): b128 lanes cover all 64 banks
#define NE 10
#define NLAYER 2
#define TOKENS 4096       // B*LAT = 8*512
#define DMODEL 256
#define DFF 1024
#define MAX_TILES 74      // 4096/64 + NE partial tiles

// Build a v16bf WMMA fragment from two aligned 16B LDS runs (-> 2x ds_load_b128)
__device__ __forceinline__ v16bf frag_ld(const bf16_t* p /*16B aligned, [k..k+7],[k+16..k+23]*/)
{
  const v8bf* p0 = (const v8bf*)__builtin_assume_aligned(p, 16);
  const v8bf* p1 = (const v8bf*)__builtin_assume_aligned(p + 16, 16);
  v8bf a = *p0, b = *p1;
  return __builtin_shufflevector(a, b, 0, 1, 2, 3, 4, 5, 6, 7,
                                 8, 9, 10, 11, 12, 13, 14, 15);
}

// ---------------------------------------------------------------------------
// Generic batched GEMM:  C[z] = act( alpha * A[z] @ B[z] + bias )  (opt accum)
//   A: f32, row-major, lda (multiple of 4; 16B-aligned rows). Staged to LDS as
//      bf16 [row][k]. B: bf16 with strides (bsk,bsn); staged TRANSPOSED [n][k]
//      so fragments for both operands are contiguous 16B runs.
//   Per-z offsets: i0 = z/zdiv, i1 = z%zdiv; off = i0*off0 + i1*off1.
//   128 threads = 4 waves; each wave owns a 16-row strip: 1 A-fragment reused
//   across 4 v_wmma_f32_16x16x32_bf16 per K-step.
// ---------------------------------------------------------------------------
__global__ __launch_bounds__(128)
void k_gemm(const float* __restrict__ A, const bf16_t* __restrict__ B,
            const float* __restrict__ bias, float* __restrict__ C,
            int M, int N, int K, int lda, long bsk, long bsn, int ldc,
            int zdiv, long a0, long a1, long b0, long b1, long c0, long c1,
            float alpha, int act, int accum)
{
  __shared__ __align__(16) bf16_t As[TM][LDK];
  __shared__ __align__(16) bf16_t Bs[TN][LDK];   // transposed: [n][k]

  int z  = blockIdx.z;
  int i0 = z / zdiv, i1 = z % zdiv;
  const float*  Ab = A + (long)i0 * a0 + (long)i1 * a1;
  const bf16_t* Bb = B + (long)i0 * b0 + (long)i1 * b1;
  float*        Cb = C + (long)i0 * c0 + (long)i1 * c1;

  int m0 = blockIdx.x * TM;
  int n0 = blockIdx.y * TN;
  int tid  = threadIdx.x;
  int wave = tid >> 5;
  int lane = tid & 31;
  int hrow = lane & 15;
  int kb   = (lane >> 4) * 8;   // K sub-block base: 0 or 8

  v8f acc[4] = {};

  for (int k0 = 0; k0 < K; k0 += TK) {
    if (lane == 0 && k0 + TK < K) {
      int pr = m0 + wave * 16;
      if (pr < M) __builtin_prefetch(&Ab[(long)pr * lda + k0 + TK], 0, 1);
    }
    // ---- stage A: 64x32 f32 -> bf16, float4 loads + packed 8B stores
    for (int i = tid; i < (TM * TK) / 4; i += 128) {
      int r  = i >> 3;          // 8 float4 per row
      int c4 = (i & 7) * 4;
      int gr = m0 + r;
      float4 v = {0.f, 0.f, 0.f, 0.f};
      if (gr < M) v = *(const float4*)&Ab[(long)gr * lda + k0 + c4];
      v4bf o; o[0] = (bf16_t)v.x; o[1] = (bf16_t)v.y; o[2] = (bf16_t)v.z; o[3] = (bf16_t)v.w;
      *(v4bf*)__builtin_assume_aligned(&As[r][c4], 8) = o;
    }
    // ---- stage B (transposed into Bs[n][k])
    if (bsk == 1 && (bsn & 3) == 0) {
      // k-contiguous (e.g. [N,K] weights): 8B loads, contiguous 8B LDS stores
      for (int i = tid; i < (TK * TN) / 4; i += 128) {
        int c  = i >> 3;        // n
        int r4 = (i & 7) * 4;   // k
        int gn = n0 + c;
        v4bf o = {};
        if (gn < N) o = *(const v4bf*)&Bb[(long)gn * bsn + (k0 + r4)];
        *(v4bf*)__builtin_assume_aligned(&Bs[c][r4], 8) = o;
      }
    } else if (bsn == 1 && (bsk & 3) == 0) {
      // n-contiguous (e.g. [K,N] weights / V): 8B loads, transpose-scatter
      for (int i = tid; i < (TK * TN) / 4; i += 128) {
        int r  = i >> 4;        // k
        int c4 = (i & 15) * 4;  // n
        int gk = k0 + r, gn = n0 + c4;
        v4bf o = {};
        if (gn < N) o = *(const v4bf*)&Bb[(long)gk * bsk + gn];
        Bs[c4 + 0][r] = o[0]; Bs[c4 + 1][r] = o[1];
        Bs[c4 + 2][r] = o[2]; Bs[c4 + 3][r] = o[3];
      }
    } else {
      for (int i = tid; i < TK * TN; i += 128) {
        int r = i >> 6, c = i & 63;
        int gk = k0 + r, gn = n0 + c;
        bf16_t v = (bf16_t)0.f;
        if (gn < N) v = Bb[(long)gk * bsk + (long)gn * bsn];
        Bs[c][r] = v;
      }
    }
    __syncthreads();

    v16bf af = frag_ld(&As[wave * 16 + hrow][kb]);
#pragma unroll
    for (int nt = 0; nt < 4; ++nt) {
      v16bf bfv = frag_ld(&Bs[nt * 16 + hrow][kb]);
      acc[nt] = __builtin_amdgcn_wmma_f32_16x16x32_bf16(
          false, af, false, bfv, (short)0, acc[nt], false, false);
    }
    __syncthreads();
  }

  int rb = m0 + wave * 16 + ((lane >> 4) * 8);
#pragma unroll
  for (int nt = 0; nt < 4; ++nt) {
    int col = n0 + nt * 16 + hrow;
    if (col >= N) continue;
    float bsv = bias ? bias[col] : 0.f;
#pragma unroll
    for (int v = 0; v < 8; ++v) {
      int row = rb + v;
      if (row >= M) continue;
      float r = acc[nt][v] * alpha + bsv;
      if (act == 1) r = fmaxf(r, 0.f);
      else if (act == 2) r = tanhf(r);
      long idx = (long)row * ldc + col;
      if (accum) r += Cb[idx];
      Cb[idx] = r;
    }
  }
}

// ---------------------------------------------------------------------------
// Grouped MoE GEMM over expert tiles (top-1 routing, only routed experts).
// stage 1: rows gathered via perm from z, B = e_w1[e] [K=256,N=1024], relu,
//          store hh[permPos].
// stage 2: rows = permPos from hh, B = e_w2[e] [K=1024,N=256], scale rmax,
//          scatter to y[perm[permPos]].
// ---------------------------------------------------------------------------
__global__ __launch_bounds__(128)
void k_moe_gemm(const float* __restrict__ A, const bf16_t* __restrict__ Bw,
                const float* __restrict__ bias, float* __restrict__ Cout,
                const int* __restrict__ perm, const int* __restrict__ tE,
                const int* __restrict__ tR, const int* __restrict__ tEnd,
                const int* __restrict__ nT, const float* __restrict__ rscale,
                int N, int K, long wstride, int stage)
{
  int tile = blockIdx.x;
  if (tile >= *nT) return;
  int e    = tE[tile];
  int row0 = tR[tile];
  int rend = tEnd[tile];
  const bf16_t* Bb = Bw + (long)e * wstride;
  int n0 = blockIdx.y * TN;

  __shared__ __align__(16) bf16_t As[TM][LDK];
  __shared__ __align__(16) bf16_t Bs[TN][LDK];
  int tid = threadIdx.x, wave = tid >> 5, lane = tid & 31;
  int hrow = lane & 15, kb = (lane >> 4) * 8;
  v8f acc[4] = {};

  for (int k0 = 0; k0 < K; k0 += TK) {
    // gathered A rows: float4 loads
    for (int i = tid; i < (TM * TK) / 4; i += 128) {
      int r  = i >> 3;
      int c4 = (i & 7) * 4;
      int pp = row0 + r;
      float4 v = {0.f, 0.f, 0.f, 0.f};
      if (pp < rend) {
        int srow = (stage == 1) ? perm[pp] : pp;
        v = *(const float4*)&A[(long)srow * K + k0 + c4];
      }
      v4bf o; o[0] = (bf16_t)v.x; o[1] = (bf16_t)v.y; o[2] = (bf16_t)v.z; o[3] = (bf16_t)v.w;
      *(v4bf*)__builtin_assume_aligned(&As[r][c4], 8) = o;
    }
    // B natural [K,N]: n-contiguous 8B loads, transpose into Bs[n][k]
    for (int i = tid; i < (TK * TN) / 4; i += 128) {
      int r  = i >> 4;
      int c4 = (i & 15) * 4;
      v4bf o = *(const v4bf*)&Bb[(long)(k0 + r) * N + (n0 + c4)];
      Bs[c4 + 0][r] = o[0]; Bs[c4 + 1][r] = o[1];
      Bs[c4 + 2][r] = o[2]; Bs[c4 + 3][r] = o[3];
    }
    __syncthreads();

    v16bf af = frag_ld(&As[wave * 16 + hrow][kb]);
#pragma unroll
    for (int nt = 0; nt < 4; ++nt) {
      v16bf bfv = frag_ld(&Bs[nt * 16 + hrow][kb]);
      acc[nt] = __builtin_amdgcn_wmma_f32_16x16x32_bf16(
          false, af, false, bfv, (short)0, acc[nt], false, false);
    }
    __syncthreads();
  }

  int rb = wave * 16 + ((lane >> 4) * 8);
#pragma unroll
  for (int nt = 0; nt < 4; ++nt) {
    int col = n0 + nt * 16 + hrow;
    float bsv = bias[(long)e * N + col];
#pragma unroll
    for (int v = 0; v < 8; ++v) {
      int pp = row0 + rb + v;
      if (pp >= rend) continue;
      float r = acc[nt][v] + bsv;
      if (stage == 1) {
        Cout[(long)pp * N + col] = fmaxf(r, 0.f);
      } else {
        int tok = perm[pp];
        Cout[(long)tok * N + col] = r * rscale[tok];
      }
    }
  }
}

// ---------------------------------------------------------------------------
// Tokenizer: conv1d(k=5, replicate pad) + LP-pool(p=2) + PE + LayerNorm.
// ---------------------------------------------------------------------------
__global__ __launch_bounds__(256)
void k_tokenize(const float* __restrict__ x, const float* __restrict__ cw,
                const float* __restrict__ cb, const float* __restrict__ pe,
                const float* __restrict__ g, const float* __restrict__ bb,
                float* __restrict__ z)
{
  int tok = blockIdx.x;
  int b = tok >> 9, t = tok & 511;
  int d = threadIdx.x;
  const float* xb = x + (long)b * 1024;
  float h0 = cb[d], h1 = cb[d];
#pragma unroll
  for (int j = 0; j < 5; ++j) {
    float w = cw[d * 5 + j];
    int p0 = 2 * t + j - 2;     p0 = p0 < 0 ? 0 : (p0 > 1023 ? 1023 : p0);
    int p1 = 2 * t + 1 + j - 2; p1 = p1 < 0 ? 0 : (p1 > 1023 ? 1023 : p1);
    h0 += xb[p0] * w;
    h1 += xb[p1] * w;
  }
  float val = sqrtf(h0 * h0 + h1 * h1) + pe[(long)t * DMODEL + d];

  __shared__ float red[256];
  red[d] = val; __syncthreads();
  for (int s = 128; s > 0; s >>= 1) { if (d < s) red[d] += red[d + s]; __syncthreads(); }
  float mu = red[0] * (1.f / 256.f); __syncthreads();
  float dv = val - mu;
  red[d] = dv * dv; __syncthreads();
  for (int s = 128; s > 0; s >>= 1) { if (d < s) red[d] += red[d + s]; __syncthreads(); }
  float var = red[0] * (1.f / 256.f);
  z[(long)tok * DMODEL + d] = dv * rsqrtf(var + 1e-5f) * g[d] + bb[d];
}

// z = LN(z + y)
__global__ __launch_bounds__(256)
void k_add_ln(float* __restrict__ z, const float* __restrict__ y,
              const float* __restrict__ g, const float* __restrict__ b)
{
  int tok = blockIdx.x, d = threadIdx.x;
  long i = (long)tok * DMODEL + d;
  float val = z[i] + y[i];
  __shared__ float red[256];
  red[d] = val; __syncthreads();
  for (int s = 128; s > 0; s >>= 1) { if (d < s) red[d] += red[d + s]; __syncthreads(); }
  float mu = red[0] * (1.f / 256.f); __syncthreads();
  float dv = val - mu;
  red[d] = dv * dv; __syncthreads();
  for (int s = 128; s > 0; s >>= 1) { if (d < s) red[d] += red[d + s]; __syncthreads(); }
  float var = red[0] * (1.f / 256.f);
  z[i] = dv * rsqrtf(var + 1e-5f) * g[d] + b[d];
}

// softmax over rows of length 512 (attention scores), in-place
__global__ __launch_bounds__(256)
void k_softmax512(float* __restrict__ s)
{
  float* p = s + (long)blockIdx.x * 512;
  int tid = threadIdx.x;
  float a = p[tid], b = p[tid + 256];
  __shared__ float red[256];
  red[tid] = fmaxf(a, b); __syncthreads();
  for (int st = 128; st > 0; st >>= 1) { if (tid < st) red[tid] = fmaxf(red[tid], red[tid + st]); __syncthreads(); }
  float m = red[0]; __syncthreads();
  float ea = expf(a - m), eb = expf(b - m);
  red[tid] = ea + eb; __syncthreads();
  for (int st = 128; st > 0; st >>= 1) { if (tid < st) red[tid] += red[tid + st]; __syncthreads(); }
  float inv = 1.f / red[0];
  p[tid] = ea * inv;
  p[tid + 256] = eb * inv;
}

// top-1 router: logits, softmax, argmax, stats. One wave per token.
__global__ __launch_bounds__(128)
void k_router(const float* __restrict__ z, const float* __restrict__ rw,
              const float* __restrict__ rb, int* __restrict__ routes,
              float* __restrict__ rmax, float* __restrict__ counts,
              float* __restrict__ rprob)
{
  int t = blockIdx.x * 4 + (threadIdx.x >> 5);
  int lane = threadIdx.x & 31;
  if (t >= TOKENS) return;
  float lg[NE];
#pragma unroll
  for (int e = 0; e < NE; ++e) {
    float s = 0.f;
    for (int d = lane; d < DMODEL; d += 32) s += z[(long)t * DMODEL + d] * rw[e * DMODEL + d];
    for (int o = 16; o > 0; o >>= 1) s += __shfl_xor(s, o, 32);
    lg[e] = s + rb[e];
  }
  float mx = lg[0]; int am = 0;
#pragma unroll
  for (int e = 1; e < NE; ++e) if (lg[e] > mx) { mx = lg[e]; am = e; }
  float den = 0.f;
#pragma unroll
  for (int e = 0; e < NE; ++e) den += expf(lg[e] - mx);
  if (lane == 0) {
    routes[t] = am;
    rmax[t] = 1.f / den;
    atomicAdd(&counts[am], 1.f);
#pragma unroll
    for (int e = 0; e < NE; ++e) atomicAdd(&rprob[e], expf(lg[e] - mx) / den);
  }
}

// build per-expert offsets + tile table (single thread; NE=10 trivial)
__global__ void k_moe_offsets(const float* __restrict__ counts, int* __restrict__ fill,
                              int* __restrict__ tE, int* __restrict__ tR,
                              int* __restrict__ tEnd, int* __restrict__ nT)
{
  int o = 0, nt = 0;
  for (int e = 0; e < NE; ++e) {
    int c = (int)(counts[e] + 0.5f);
    fill[e] = o;
    int ntile = (c + 63) >> 6;
    for (int i = 0; i < ntile; ++i) { tE[nt] = e; tR[nt] = o + i * 64; tEnd[nt] = o + c; ++nt; }
    o += c;
  }
  *nT = nt;
}

__global__ __launch_bounds__(256)
void k_moe_scatter(const int* __restrict__ routes, int* __restrict__ fill,
                   int* __restrict__ perm)
{
  int t = blockIdx.x * 256 + threadIdx.x;
  if (t < TOKENS) {
    int p = atomicAdd(&fill[routes[t]], 1);
    perm[p] = t;
  }
}

__global__ __launch_bounds__(256)
void k_cast(const float* __restrict__ s, bf16_t* __restrict__ d, long n)
{
  for (long i = blockIdx.x * 256L + threadIdx.x; i < n; i += (long)gridDim.x * 256)
    d[i] = (bf16_t)s[i];
}

// shifted copies of z for the 4-tap deconv GEMM decomposition
__global__ __launch_bounds__(256)
void k_shift(const float* __restrict__ z, float* __restrict__ zp, float* __restrict__ zn)
{
  long i = blockIdx.x * 256L + threadIdx.x;
  if (i >= (long)TOKENS * DMODEL) return;
  long tok = i >> 8; int d = (int)(i & 255);
  int t = (int)(tok & 511);
  zp[i] = (t > 0)   ? z[(tok - 1) * DMODEL + d] : 0.f;
  zn[i] = (t < 511) ? z[(tok + 1) * DMODEL + d] : 0.f;
}

// final 128->1 projection, one wave per output sample
__global__ __launch_bounds__(256)
void k_lin2(const float* __restrict__ u1, const float* __restrict__ w,
            const float* __restrict__ b, float* __restrict__ out)
{
  int row = blockIdx.x * 8 + (threadIdx.x >> 5);
  int lane = threadIdx.x & 31;
  float s = 0.f;
  for (int i = lane; i < 128; i += 32) s += u1[(long)row * 128 + i] * w[i];
  for (int o = 16; o > 0; o >>= 1) s += __shfl_xor(s, o, 32);
  if (lane == 0) out[row] = s + b[0];
}

__global__ void k_loss(const float* __restrict__ counts, const float* __restrict__ rprob,
                       float* __restrict__ out)
{
  float l = 0.f;
  for (int i = 0; i < NLAYER * NE; ++i)
    l += (counts[i] * (1.f / 4096.f)) * (rprob[i] * (1.f / 4096.f));
  out[0] = (float)NE * l;
}

// ---------------------------------------------------------------------------
static void gemm_launch(hipStream_t st, const float* A, const bf16_t* B,
                        const float* bias, float* C,
                        int M, int N, int K, int lda, long bsk, long bsn, int ldc,
                        int gz, int zdiv, long a0, long a1, long b0, long b1,
                        long c0, long c1, float alpha, int act, int accum)
{
  dim3 grid((M + TM - 1) / TM, (N + TN - 1) / TN, gz);
  k_gemm<<<grid, 128, 0, st>>>(A, B, bias, C, M, N, K, lda, bsk, bsn, ldc,
                               zdiv, a0, a1, b0, b1, c0, c1, alpha, act, accum);
}

extern "C" void kernel_launch(void* const* d_in, const int* in_sizes, int n_in,
                              void* d_out, int out_size, void* d_ws, size_t ws_size,
                              hipStream_t stream)
{
  (void)in_sizes; (void)n_in; (void)out_size; (void)ws_size;
  const float* x        = (const float*)d_in[0];
  const float* conv_w   = (const float*)d_in[1];
  const float* conv_b   = (const float*)d_in[2];
  const float* pe       = (const float*)d_in[3];
  const float* ln0_g    = (const float*)d_in[4];
  const float* ln0_b    = (const float*)d_in[5];
  const float* wq       = (const float*)d_in[6];
  const float* bq       = (const float*)d_in[7];
  const float* wk       = (const float*)d_in[8];
  const float* bk       = (const float*)d_in[9];
  const float* wv       = (const float*)d_in[10];
  const float* bv       = (const float*)d_in[11];
  const float* wo       = (const float*)d_in[12];
  const float* bo       = (const float*)d_in[13];
  const float* ln1_g    = (const float*)d_in[14];
  const float* ln1_b    = (const float*)d_in[15];
  const float* ln2_g    = (const float*)d_in[16];
  const float* ln2_b    = (const float*)d_in[17];
  const float* router_w = (const float*)d_in[18];
  const float* router_b = (const float*)d_in[19];
  const float* e_w1     = (const float*)d_in[20];
  const float* e_b1     = (const float*)d_in[21];
  const float* e_w2     = (const float*)d_in[22];
  const float* e_b2     = (const float*)d_in[23];
  const float* deconv_w = (const float*)d_in[24];
  const float* deconv_b = (const float*)d_in[25];
  const float* lin1_w   = (const float*)d_in[26];
  const float* lin1_b   = (const float*)d_in[27];
  const float* lin2_w   = (const float*)d_in[28];
  const float* lin2_b   = (const float*)d_in[29];
  float* out = (float*)d_out;

  // workspace carving
  char* cur = (char*)d_ws;
  auto alloc = [&](size_t n) -> char* {
    char* p = cur; cur += (n + 255) & ~(size_t)255; return p;
  };
  float* z      = (float*)alloc((size_t)TOKENS * DMODEL * 4);
  float* q      = (float*)alloc((size_t)TOKENS * DMODEL * 4);
  float* kk     = (float*)alloc((size_t)TOKENS * DMODEL * 4);
  float* vv     = (float*)alloc((size_t)TOKENS * DMODEL * 4);
  float* y      = (float*)alloc((size_t)TOKENS * DMODEL * 4);
  float* aout   = (float*)alloc((size_t)TOKENS * DMODEL * 4);
  float* zprev  = (float*)alloc((size_t)TOKENS * DMODEL * 4);
  float* znext  = (float*)alloc((size_t)TOKENS * DMODEL * 4);
  float* scores = (float*)alloc((size_t)16 * 512 * 512 * 4);   // [B*H,512,512]
  float* hh     = (float*)alloc((size_t)TOKENS * DFF * 4);
  float* u      = (float*)alloc((size_t)8192 * DMODEL * 4);
  float* u1     = (float*)alloc((size_t)8192 * 128 * 4);
  bf16_t* kbf   = (bf16_t*)alloc((size_t)TOKENS * DMODEL * 2);
  bf16_t* vbf   = (bf16_t*)alloc((size_t)TOKENS * DMODEL * 2);
  bf16_t* wqbf  = (bf16_t*)alloc((size_t)NLAYER * DMODEL * DMODEL * 2);
  bf16_t* wkbf  = (bf16_t*)alloc((size_t)NLAYER * DMODEL * DMODEL * 2);
  bf16_t* wvbf  = (bf16_t*)alloc((size_t)NLAYER * DMODEL * DMODEL * 2);
  bf16_t* wobf  = (bf16_t*)alloc((size_t)NLAYER * DMODEL * DMODEL * 2);
  bf16_t* ew1bf = (bf16_t*)alloc((size_t)NLAYER * NE * DMODEL * DFF * 2);
  bf16_t* ew2bf = (bf16_t*)alloc((size_t)NLAYER * NE * DFF * DMODEL * 2);
  bf16_t* dwbf  = (bf16_t*)alloc((size_t)DMODEL * DMODEL * 4 * 2);
  bf16_t* l1bf  = (bf16_t*)alloc((size_t)128 * DMODEL * 2);
  float* stats  = (float*)alloc((size_t)2 * NLAYER * NE * 4);  // counts | rprob
  int* routes   = (int*)alloc((size_t)TOKENS * 4);
  float* rmax   = (float*)alloc((size_t)TOKENS * 4);
  int* perm     = (int*)alloc((size_t)TOKENS * 4);
  int* fill     = (int*)alloc((size_t)NE * 4);
  int* tE       = (int*)alloc((size_t)MAX_TILES * 4);
  int* tR       = (int*)alloc((size_t)MAX_TILES * 4);
  int* tEnd     = (int*)alloc((size_t)MAX_TILES * 4);
  int* nT       = (int*)alloc(4);
  float* counts = stats;
  float* rprob  = stats + NLAYER * NE;

  auto cast = [&](const float* s, bf16_t* d, long n) {
    int blocks = (int)((n + 255) / 256); if (blocks > 8192) blocks = 8192;
    k_cast<<<blocks, 256, 0, stream>>>(s, d, n);
  };

  // pre-cast all GEMM weights to bf16 (fit in L2; one pass each)
  cast(wq, wqbf, (long)NLAYER * DMODEL * DMODEL);
  cast(wk, wkbf, (long)NLAYER * DMODEL * DMODEL);
  cast(wv, wvbf, (long)NLAYER * DMODEL * DMODEL);
  cast(wo, wobf, (long)NLAYER * DMODEL * DMODEL);
  cast(e_w1, ew1bf, (long)NLAYER * NE * DMODEL * DFF);
  cast(e_w2, ew2bf, (long)NLAYER * NE * DFF * DMODEL);
  cast(deconv_w, dwbf, (long)DMODEL * DMODEL * 4);
  cast(lin1_w, l1bf, (long)128 * DMODEL);
  hipMemsetAsync(stats, 0, (size_t)2 * NLAYER * NE * 4, stream);

  // tokenizer: conv + LP-pool + PE + LN0
  k_tokenize<<<TOKENS, 256, 0, stream>>>(x, conv_w, conv_b, pe, ln0_g, ln0_b, z);

  const float SCALE = 0.08838834764831845f;  // 1/sqrt(128)

  for (int l = 0; l < NLAYER; ++l) {
    long wofs = (long)l * DMODEL * DMODEL;
    // Q, K, V projections: [4096,256] = z @ W^T  (B(k,n)=W[n*256+k], bsk==1)
    gemm_launch(stream, z, wqbf + wofs, bq + l * DMODEL, q, TOKENS, DMODEL, DMODEL,
                DMODEL, 1, DMODEL, DMODEL, 1, 1, 0,0,0,0,0,0, 1.f, 0, 0);
    gemm_launch(stream, z, wkbf + wofs, bk + l * DMODEL, kk, TOKENS, DMODEL, DMODEL,
                DMODEL, 1, DMODEL, DMODEL, 1, 1, 0,0,0,0,0,0, 1.f, 0, 0);
    gemm_launch(stream, z, wvbf + wofs, bv + l * DMODEL, vv, TOKENS, DMODEL, DMODEL,
                DMODEL, 1, DMODEL, DMODEL, 1, 1, 0,0,0,0,0,0, 1.f, 0, 0);
    cast(kk, kbf, (long)TOKENS * DMODEL);
    cast(vv, vbf, (long)TOKENS * DMODEL);

    // scores[b,h] = Q K^T / sqrt(dk): batched over z=b*2+h (zdiv=H=2)
    gemm_launch(stream, q, kbf, nullptr, scores, 512, 512, 128,
                DMODEL, 1, DMODEL, 512,
                16, 2, /*a*/131072, 128, /*b*/131072, 128, /*c*/524288, 262144,
                SCALE, 0, 0);
    k_softmax512<<<8192, 256, 0, stream>>>(scores);
    // attn @ V -> aout[b,h]  (bsn==1 path)
    gemm_launch(stream, scores, vbf, nullptr, aout, 512, 128, 512,
                512, DMODEL, 1, DMODEL,
                16, 2, /*a*/524288, 262144, /*b*/131072, 128, /*c*/131072, 128,
                1.f, 0, 0);
    // output projection
    gemm_launch(stream, aout, wobf + wofs, bo + l * DMODEL, y, TOKENS, DMODEL, DMODEL,
                DMODEL, 1, DMODEL, DMODEL, 1, 1, 0,0,0,0,0,0, 1.f, 0, 0);
    k_add_ln<<<TOKENS, 256, 0, stream>>>(z, y, ln1_g + l * DMODEL, ln1_b + l * DMODEL);

    // Switch FFN: route, gather, grouped WMMA GEMMs, scatter
    k_router<<<TOKENS / 4, 128, 0, stream>>>(z, router_w + (long)l * NE * DMODEL,
                                             router_b + l * NE, routes, rmax,
                                             counts + l * NE, rprob + l * NE);
    k_moe_offsets<<<1, 1, 0, stream>>>(counts + l * NE, fill, tE, tR, tEnd, nT);
    k_moe_scatter<<<TOKENS / 256, 256, 0, stream>>>(routes, fill, perm);
    k_moe_gemm<<<dim3(MAX_TILES, DFF / TN, 1), 128, 0, stream>>>(
        z, ew1bf + (long)l * NE * DMODEL * DFF, e_b1 + (long)l * NE * DFF, hh,
        perm, tE, tR, tEnd, nT, nullptr, DFF, DMODEL, (long)DMODEL * DFF, 1);
    k_moe_gemm<<<dim3(MAX_TILES, DMODEL / TN, 1), 128, 0, stream>>>(
        hh, ew2bf + (long)l * NE * DFF * DMODEL, e_b2 + (long)l * NE * DMODEL, y,
        perm, tE, tR, tEnd, nT, rmax, DMODEL, DFF, (long)DFF * DMODEL, 2);
    k_add_ln<<<TOKENS, 256, 0, stream>>>(z, y, ln2_g + l * DMODEL, ln2_b + l * DMODEL);
  }

  // ConvTranspose1d(k=4,s=2,p=1) as 4 accumulate-GEMMs on shifted inputs.
  // deconv_w[cin,cout,kk]: bsk=DMODEL*4, bsn=4 (generic path), base offset kk.
  k_shift<<<TOKENS, 256, 0, stream>>>(z, zprev, znext);
  // even outputs u[2m] = z[m] W[:,:,1] + z[m-1] W[:,:,3]  (rows interleave via ldc=512)
  gemm_launch(stream, z,     dwbf + 1, deconv_b, u, 512, DMODEL, DMODEL,
              DMODEL, DMODEL * 4, 4, 512, 8, 1, 131072, 0, 0, 0, 262144, 0, 1.f, 0, 0);
  gemm_launch(stream, zprev, dwbf + 3, nullptr,  u, 512, DMODEL, DMODEL,
              DMODEL, DMODEL * 4, 4, 512, 8, 1, 131072, 0, 0, 0, 262144, 0, 1.f, 0, 1);
  // odd outputs u[2m+1] = z[m+1] W[:,:,0] + z[m] W[:,:,2]
  gemm_launch(stream, znext, dwbf + 0, deconv_b, u + DMODEL, 512, DMODEL, DMODEL,
              DMODEL, DMODEL * 4, 4, 512, 8, 1, 131072, 0, 0, 0, 262144, 0, 1.f, 0, 0);
  gemm_launch(stream, z,     dwbf + 2, nullptr,  u + DMODEL, 512, DMODEL, DMODEL,
              DMODEL, DMODEL * 4, 4, 512, 8, 1, 131072, 0, 0, 0, 262144, 0, 1.f, 0, 1);

  // head: tanh(u @ lin1^T + b1) then 128->1
  gemm_launch(stream, u, l1bf, lin1_b, u1, 8192, 128, DMODEL,
              DMODEL, 1, DMODEL, 128, 1, 1, 0,0,0,0,0,0, 1.f, 2, 0);
  k_lin2<<<8192 / 8, 256, 0, stream>>>(u1, lin2_w, lin2_b, out);
  k_loss<<<1, 1, 0, stream>>>(counts, rprob, out + 8192);
}
